// AttentionAGaLiTeLayer_25993142076006
// MI455X (gfx1250) — matrix operand order repaired and checked by reading, not compiled, not run on hardware
//
#include <hip/hip_runtime.h>
#include <math.h>

// ---- problem constants --------------------------------------------------
#define T_DIM   1024
#define B_DIM   8
#define DIN     512
#define H_DIM   8
#define HD_DIM  64
#define ETA_DIM 4
#define R_DIM   4
#define E_DIM   256            // HD*ETA
#define EPS_V   1e-5f
#define M_TOT   (T_DIM * B_DIM)      // 8192 rows for all GEMMs
#define N_KQV   (H_DIM * HD_DIM * 5) // 2560
#define N_P     (H_DIM * ETA_DIM * 3) // 96
#define N_PROJ  DIN                  // 512
#define BH_TOT  (B_DIM * H_DIM)      // 64
#define NCHUNK  64
#define CLEN    (T_DIM / NCHUNK)     // 16

#define BT_STRIDE 72   // padded LDS row stride (ushorts) to break bank conflicts

// ---- CDNA5 async global->LDS staging (guarded) ---------------------------
#if defined(__has_builtin)
# if __has_builtin(__builtin_amdgcn_global_load_async_to_lds_b128)
#  define HAVE_ASYNC_LDS 1
# endif
#endif
#ifndef HAVE_ASYNC_LDS
# define HAVE_ASYNC_LDS 0
#endif

typedef __attribute__((ext_vector_type(16))) __bf16          v16bf;
typedef __attribute__((ext_vector_type(16))) unsigned short  v16u;
typedef __attribute__((ext_vector_type(8)))  float           v8f;

#if HAVE_ASYNC_LDS
// builtin expects gcc-style vector_size int4 pointers, non-const
typedef int v4i_vs __attribute__((vector_size(4 * sizeof(int))));
typedef v4i_vs __attribute__((address_space(1)))* gv4i_p;   // global src
typedef v4i_vs __attribute__((address_space(3)))* lv4i_p;   // LDS dst
#endif

__device__ __forceinline__ unsigned short f32_to_bf16_rne(float f) {
  unsigned int u = __float_as_uint(f);
  u += 0x7FFFu + ((u >> 16) & 1u);   // round-to-nearest-even
  return (unsigned short)(u >> 16);
}

__device__ __forceinline__ float sigm(float x) {
  return 1.0f / (1.0f + __expf(-x));
}

// omegas = linspace(-pi, pi, 4)
#define OM0 (-3.14159265358979323846f)
#define DOM (2.0f * 3.14159265358979323846f / 3.0f)

// ---- fp32 -> bf16 conversion (grid-stride) ------------------------------
__global__ void cvt_f32_to_bf16(const float* __restrict__ src,
                                unsigned short* __restrict__ dst, int n) {
  int i = blockIdx.x * blockDim.x + threadIdx.x;
  int stride = gridDim.x * blockDim.x;
  for (; i < n; i += stride) dst[i] = f32_to_bf16_rne(src[i]);
}

// ---- A fragment load: 16x32 bf16 per ISA 7.12.2 layout -------------------
__device__ __forceinline__ v16bf load_a_frag(const unsigned short* __restrict__ arow,
                                             int kk, int khalf) {
  v16u au;
#pragma unroll
  for (int v = 0; v < 8; ++v) {
    int k = kk + ((v & 4) ? 16 : 0) + khalf + 2 * (v & 3);
    au[2 * v]     = arow[k];
    au[2 * v + 1] = arow[k + 1];
  }
  return __builtin_bit_cast(v16bf, au);
}

// ---- B fragment load from staged LDS tile --------------------------------
__device__ __forceinline__ v16bf load_b_frag(const unsigned short* bt, int j) {
  v16u bu;
#pragma unroll
  for (int v = 0; v < 16; ++v) bu[v] = bt[(v + (v & 8)) * BT_STRIDE + j * 16];   // K = 8h + v + (v&8), N = j*16 + lane%16
  return __builtin_bit_cast(v16bf, bu);
}

// ---- 16B stage: async global->LDS when available, else via VGPRs ---------
__device__ __forceinline__ void stage16(unsigned short* ldst,
                                        const unsigned short* gsrc) {
#if HAVE_ASYNC_LDS
  __builtin_amdgcn_global_load_async_to_lds_b128((gv4i_p)gsrc, (lv4i_p)ldst, 0, 0);
#else
  *(uint4*)ldst = *(const uint4*)gsrc;
#endif
}

__device__ __forceinline__ void stage_wait() {
#if HAVE_ASYNC_LDS
# if defined(__has_builtin) && __has_builtin(__builtin_amdgcn_s_wait_asynccnt)
  __builtin_amdgcn_s_wait_asynccnt(0);
# else
  asm volatile("s_wait_asynccnt 0" ::: "memory");
# endif
#endif
}

// ---- tiled bf16 WMMA GEMM with LDS-staged, double-buffered B -------------
// Block: 256 threads = 8 waves; block tile 128(M) x 64(N); K-step 32.
// Requires M%128==0, N%64==0, K%32==0.
__global__ void gemm_bf16_wmma_tiled(const unsigned short* __restrict__ A,
                                     const unsigned short* __restrict__ Bw,
                                     const float* __restrict__ bias,
                                     float* __restrict__ C,
                                     int M, int N, int K) {
  __shared__ unsigned short btile[2][32 * BT_STRIDE];

  const int tid     = threadIdx.x;
  const int lane    = tid & 31;
  const int wave    = tid >> 5;
  const int colBase = blockIdx.x * 64;
  const int rowBase = blockIdx.y * 128 + wave * 16;
  const int m       = rowBase + (lane & 15);
  const int khalf   = (lane < 16) ? 0 : 8;
  const unsigned short* arow = A + (size_t)m * K;

  // staging role: thread tid moves one 16B chunk of the 32x64 B tile
  const int srow = tid >> 3;   // 0..31 (K within tile)
  const int sseg = tid & 7;    // 0..7  (8 ushorts each)
  const int soff = srow * BT_STRIDE + sseg * 8;

  v8f acc0 = {}, acc1 = {}, acc2 = {}, acc3 = {};

  // prologue: stage first B tile
  stage16(&btile[0][soff], Bw + (size_t)srow * N + colBase + sseg * 8);

  int p = 0;
  for (int kk = 0; kk < K; kk += 32) {
    stage_wait();                         // previous async stage complete
    __syncthreads();                      // buffer p visible to all waves
    if (kk + 32 < K) {                    // stage next tile into p^1
      const unsigned short* gsrc = Bw + (size_t)(kk + 32 + srow) * N + colBase + sseg * 8;
      __builtin_prefetch(gsrc, 0, 3);
      stage16(&btile[p ^ 1][soff], gsrc);
    }
    const v16bf af = load_a_frag(arow, kk, khalf);
    const unsigned short* bt = &btile[p][(lane >> 4) * 8 * BT_STRIDE + (lane & 15)];  // K base = 8*(lane/16), N = lane%16
    // load all four B fragments, then pin them live so the four WMMAs issue
    // back-to-back behind a single s_wait instead of one wait per WMMA.
    v16bf b0 = load_b_frag(bt, 0);
    v16bf b1 = load_b_frag(bt, 1);
    v16bf b2 = load_b_frag(bt, 2);
    v16bf b3 = load_b_frag(bt, 3);
    asm volatile("" : "+v"(b0), "+v"(b1), "+v"(b2), "+v"(b3));
    acc0 = __builtin_amdgcn_wmma_f32_16x16x32_bf16(false, af, false, b0, (short)0, acc0, false, false);
    acc1 = __builtin_amdgcn_wmma_f32_16x16x32_bf16(false, af, false, b1, (short)0, acc1, false, false);
    acc2 = __builtin_amdgcn_wmma_f32_16x16x32_bf16(false, af, false, b2, (short)0, acc2, false, false);
    acc3 = __builtin_amdgcn_wmma_f32_16x16x32_bf16(false, af, false, b3, (short)0, acc3, false, false);
    p ^= 1;
  }

  // C/D layout: VGPR i -> row = rowBase + i + (lane<16?0:8), col = lane%16
  const int rb = rowBase + ((lane < 16) ? 0 : 8);
  v8f accs[4] = {acc0, acc1, acc2, acc3};
#pragma unroll
  for (int j = 0; j < 4; ++j) {
    const int col = colBase + j * 16 + (lane & 15);
    const float bv = bias ? bias[col] : 0.0f;
#pragma unroll
    for (int i = 0; i < 8; ++i) C[(size_t)(rb + i) * N + col] = accs[j][i] + bv;
  }
}

// ---- simple per-wave 16x16 WMMA GEMM (small N not divisible by 64) -------
__global__ void gemm_bf16_wmma_small(const unsigned short* __restrict__ A,
                                     const unsigned short* __restrict__ Bw,
                                     const float* __restrict__ bias,
                                     float* __restrict__ C,
                                     int M, int N, int K) {
  const int lane    = threadIdx.x & 31;
  const int wave    = threadIdx.x >> 5;
  const int colBase = blockIdx.x * 16;
  const int rowBase = blockIdx.y * 128 + wave * 16;
  const int m       = rowBase + (lane & 15);
  const int khalf   = (lane < 16) ? 0 : 8;
  const unsigned short* arow = A + (size_t)m * K;

  v8f acc = {};
  for (int kk = 0; kk < K; kk += 32) {
    if (kk + 64 < K) __builtin_prefetch(arow + kk + 64, 0, 3);
    const v16bf af = load_a_frag(arow, kk, khalf);
    v16u bu;
    const unsigned short* bp = Bw + (size_t)(kk + (lane >> 4) * 8) * N + colBase + (lane & 15);
#pragma unroll
    for (int v = 0; v < 16; ++v) bu[v] = bp[(size_t)(v + (v & 8)) * N];
    v16bf bf = __builtin_bit_cast(v16bf, bu);
    acc = __builtin_amdgcn_wmma_f32_16x16x32_bf16(
        false, af, false, bf, (short)0, acc, false, false);
  }
  const int col = colBase + (lane & 15);
  const int rb  = rowBase + ((lane < 16) ? 0 : 8);
  const float bv = bias ? bias[col] : 0.0f;
#pragma unroll
  for (int i = 0; i < 8; ++i) C[(size_t)(rb + i) * N + col] = acc[i] + bv;
}

// =========================================================================
// Chunked parallel scan:  c_t = x_t + d_t * c_{t-1}
//   chunk summary: carry_out = X_c + D_c * carry_in  (X = local scan, D = prod d)
// =========================================================================

struct StepFeat {
  float kg, dg, q;                 // key-gated, gamma-discount, query (per e)
  float vb, db;                    // value*beta, beta-discount (per rv,dv)
  float oc0, oc1, oc2, oc3, ocv;   // oscillators
};

__device__ __forceinline__ StepFeat step_features(
    const float* __restrict__ kqv, const float* __restrict__ pfeat,
    const int* __restrict__ term, float tickb,
    int t, int b, int h, int de, int ne, int dv, int rv) {
  const float* kq = kqv   + (size_t)(((t * B_DIM + b) * H_DIM + h) * 5) * HD_DIM;
  const float* pp = pfeat + (size_t)(((t * B_DIM + b) * H_DIM + h) * 3) * ETA_DIM;
  const float notTerm = 1.0f - (float)term[t * B_DIM + b];

  StepFeat f;
  const float kraw = kq[de];
  const float qraw = kq[HD_DIM + de];
  const float graw = kq[4 * HD_DIM + de];
  const float p1 = pp[ne], p2 = pp[ETA_DIM + ne], p3 = pp[2 * ETA_DIM + ne];
  const float key = fmaxf(kraw, 0.0f) * fmaxf(p1, 0.0f);
  f.q  = fmaxf(qraw, 0.0f) * fmaxf(p2, 0.0f);
  const float gam = sigm(graw) * sigm(p3);
  f.kg = key * gam;
  f.dg = (1.0f - gam) * notTerm;

  const float tk = tickb + (float)(t + 1);
  f.oc0 = __cosf(tk * OM0);
  f.oc1 = __cosf(tk * (OM0 + DOM));
  f.oc2 = __cosf(tk * (OM0 + 2.0f * DOM));
  f.oc3 = __cosf(tk * (OM0 + 3.0f * DOM));
  f.ocv = __cosf(tk * (OM0 + DOM * (float)rv));

  const float vraw = kq[2 * HD_DIM + dv];
  const float beta = sigm(kq[3 * HD_DIM + dv]);
  f.vb = vraw * beta;
  f.db = (1.0f - beta) * notTerm;
  return f;
}

// ---- Phase A: per-chunk local scans + discount products (no barriers) ----
__global__ void scan_phaseA(const float* __restrict__ kqv,
                            const float* __restrict__ pfeat,
                            const int*   __restrict__ term,
                            const float* __restrict__ tick,
                            float* __restrict__ A_Dg,  // [NCHUNK,BH,E]
                            float* __restrict__ A_Xs,  // [NCHUNK,BH,E]
                            float* __restrict__ A_Xk,  // [NCHUNK,BH,R,E]
                            float* __restrict__ A_Db,  // [NCHUNK,BH,256]
                            float* __restrict__ A_Xv) {// [NCHUNK,BH,256]
  const int c  = blockIdx.x;
  const int bh = blockIdx.y;
  const int b  = bh / H_DIM, h = bh % H_DIM;
  const int tid = threadIdx.x;
  const int de = tid >> 2, ne = tid & 3;
  const int rv = tid >> 6, dv = tid & 63;
  const float tickb = tick[b];

  float Dg = 1.0f, Xs = 0.0f, Xk0 = 0.0f, Xk1 = 0.0f, Xk2 = 0.0f, Xk3 = 0.0f;
  float Db = 1.0f, Xv = 0.0f;

  for (int tt = 0; tt < CLEN; ++tt) {
    const int t = c * CLEN + tt;
    const StepFeat f = step_features(kqv, pfeat, term, tickb, t, b, h, de, ne, dv, rv);
    Xs  = fmaf(f.dg, Xs, f.kg);
    Xk0 = fmaf(f.dg, Xk0, f.kg * f.oc0);
    Xk1 = fmaf(f.dg, Xk1, f.kg * f.oc1);
    Xk2 = fmaf(f.dg, Xk2, f.kg * f.oc2);
    Xk3 = fmaf(f.dg, Xk3, f.kg * f.oc3);
    Dg *= f.dg;
    Xv  = fmaf(f.db, Xv, f.vb * f.ocv);
    Db *= f.db;
  }
  const size_t base = (size_t)c * BH_TOT + bh;
  A_Dg[base * E_DIM + tid] = Dg;
  A_Xs[base * E_DIM + tid] = Xs;
  A_Xk[(base * R_DIM + 0) * E_DIM + tid] = Xk0;
  A_Xk[(base * R_DIM + 1) * E_DIM + tid] = Xk1;
  A_Xk[(base * R_DIM + 2) * E_DIM + tid] = Xk2;
  A_Xk[(base * R_DIM + 3) * E_DIM + tid] = Xk3;
  A_Db[base * 256 + tid] = Db;
  A_Xv[base * 256 + tid] = Xv;
}

// ---- Phase B: sequential scan over chunk summaries (64 steps, tiny) ------
__global__ void scan_phaseB(const float* __restrict__ A_Dg,
                            const float* __restrict__ A_Xs,
                            const float* __restrict__ A_Xk,
                            const float* __restrict__ A_Db,
                            const float* __restrict__ A_Xv,
                            const float* __restrict__ tkprev,
                            const float* __restrict__ tvprev,
                            const float* __restrict__ sprev,
                            const float* __restrict__ tick,
                            float* __restrict__ Cin_k,  // [NCHUNK,BH,R,E]
                            float* __restrict__ Cin_s,  // [NCHUNK,BH,E]
                            float* __restrict__ Cin_v,  // [NCHUNK,BH,256]
                            float* __restrict__ out_fk,
                            float* __restrict__ out_fv,
                            float* __restrict__ out_fs,
                            float* __restrict__ out_tick) {
  const int bh = blockIdx.x;
  const int b  = bh / H_DIM, h = bh % H_DIM;
  const int tid = threadIdx.x;
  const int rv = tid >> 6, dv = tid & 63;

  float kc[R_DIM];
#pragma unroll
  for (int r = 0; r < R_DIM; ++r)
    kc[r] = tkprev[((size_t)(b * R_DIM + r) * H_DIM + h) * E_DIM + tid];
  float sc = sprev[((size_t)b * H_DIM + h) * E_DIM + tid];
  float vc = tvprev[((size_t)(b * R_DIM + rv) * H_DIM + h) * HD_DIM + dv];

  for (int c = 0; c < NCHUNK; ++c) {
    const size_t base = (size_t)c * BH_TOT + bh;
    Cin_s[base * E_DIM + tid] = sc;
    Cin_v[base * 256 + tid]   = vc;
#pragma unroll
    for (int r = 0; r < R_DIM; ++r) Cin_k[(base * R_DIM + r) * E_DIM + tid] = kc[r];
    const float Dg = A_Dg[base * E_DIM + tid];
    const float Db = A_Db[base * 256 + tid];
    sc = fmaf(Dg, sc, A_Xs[base * E_DIM + tid]);
    vc = fmaf(Db, vc, A_Xv[base * 256 + tid]);
#pragma unroll
    for (int r = 0; r < R_DIM; ++r)
      kc[r] = fmaf(Dg, kc[r], A_Xk[(base * R_DIM + r) * E_DIM + tid]);
  }
  // final recurrent states
  out_fs[((size_t)b * H_DIM + h) * E_DIM + tid] = sc;
#pragma unroll
  for (int r = 0; r < R_DIM; ++r)
    out_fk[((size_t)(b * R_DIM + r) * H_DIM + h) * E_DIM + tid] = kc[r];
  out_fv[((size_t)(b * R_DIM + rv) * H_DIM + h) * HD_DIM + dv] = vc;
  if (h == 0 && tid == 0) out_tick[b] = tick[b] + (float)T_DIM;
}

// ---- Phase C: replay chunks from exact carry-in; attention core ----------
__global__ void scan_phaseC(const float* __restrict__ kqv,
                            const float* __restrict__ pfeat,
                            const int*   __restrict__ term,
                            const float* __restrict__ tick,
                            const float* __restrict__ Cin_k,
                            const float* __restrict__ Cin_s,
                            const float* __restrict__ Cin_v,
                            unsigned short* __restrict__ attn_bf16) { // [T,B,512]
  const int c  = blockIdx.x;
  const int bh = blockIdx.y;
  const int b  = bh / H_DIM, h = bh % H_DIM;
  const int tid  = threadIdx.x;
  const int lane = tid & 31;
  const int wave = tid >> 5;
  const int de = tid >> 2, ne = tid & 3;
  const int rv = tid >> 6, dv = tid & 63;
  const float tickb = tick[b];

  __shared__ float red[5 * 8];
  __shared__ float tot[5];
  __shared__ float kvpart[256];

  const size_t base = (size_t)c * BH_TOT + bh;
  float kc0 = Cin_k[(base * R_DIM + 0) * E_DIM + tid];
  float kc1 = Cin_k[(base * R_DIM + 1) * E_DIM + tid];
  float kc2 = Cin_k[(base * R_DIM + 2) * E_DIM + tid];
  float kc3 = Cin_k[(base * R_DIM + 3) * E_DIM + tid];
  float sc  = Cin_s[base * E_DIM + tid];
  float vc  = Cin_v[base * 256 + tid];

  for (int tt = 0; tt < CLEN; ++tt) {
    const int t = c * CLEN + tt;
    const StepFeat f = step_features(kqv, pfeat, term, tickb, t, b, h, de, ne, dv, rv);
    sc  = fmaf(f.dg, sc, f.kg);
    kc0 = fmaf(f.dg, kc0, f.kg * f.oc0);
    kc1 = fmaf(f.dg, kc1, f.kg * f.oc1);
    kc2 = fmaf(f.dg, kc2, f.kg * f.oc2);
    kc3 = fmaf(f.dg, kc3, f.kg * f.oc3);
    vc  = fmaf(f.db, vc, f.vb * f.ocv);

    // reductions over E: kdq[0..3] and norm
    float vals[5] = {kc0 * f.q, kc1 * f.q, kc2 * f.q, kc3 * f.q, sc * f.q};
#pragma unroll
    for (int qi = 0; qi < 5; ++qi) {
      float v = vals[qi];
      for (int off = 16; off > 0; off >>= 1) v += __shfl_down(v, off, 32);
      if (lane == 0) red[qi * 8 + wave] = v;
    }
    __syncthreads();
    if (tid < 5) {
      float s = 0.0f;
#pragma unroll
      for (int i = 0; i < 8; ++i) s += red[tid * 8 + i];
      tot[tid] = s;
    }
    __syncthreads();
    const float norm = tot[4];
    kvpart[tid] = vc * tot[rv];            // final_values * kdq (per r,d)
    __syncthreads();
    if (tid < 64) {
      const float kv = kvpart[tid] + kvpart[64 + tid] +
                       kvpart[128 + tid] + kvpart[192 + tid];
      const float a = kv / (2.0f * (float)R_DIM * norm + EPS_V);
      attn_bf16[(size_t)(t * B_DIM + b) * (H_DIM * HD_DIM) + h * HD_DIM + tid] =
          f32_to_bf16_rne(a);
    }
    __syncthreads();   // protect red/kvpart before next step
  }
}

// ---- host-side orchestration -------------------------------------------
extern "C" void kernel_launch(void* const* d_in, const int* in_sizes, int n_in,
                              void* d_out, int out_size, void* d_ws, size_t ws_size,
                              hipStream_t stream) {
  const float* inputs = (const float*)d_in[0];   // [T,B,DIN]
  const int*   term   = (const int*)  d_in[1];   // [T,B]
  const float* tkprev = (const float*)d_in[2];   // [B,R,H,E]
  const float* tvprev = (const float*)d_in[3];   // [B,R,H,HD]
  const float* sprev  = (const float*)d_in[4];   // [B,H,E]
  const float* tick   = (const float*)d_in[5];   // [B,1]
  const float* Wkqv   = (const float*)d_in[6];   // [DIN, N_KQV]
  const float* bkqv   = (const float*)d_in[7];
  const float* Wp     = (const float*)d_in[8];   // [DIN, N_P]
  const float* bp     = (const float*)d_in[9];
  const float* Wproj  = (const float*)d_in[10];  // [DIN, DIN]
  const float* bproj  = (const float*)d_in[11];

  // workspace carve-up (256B aligned)
  char* ws = (char*)d_ws;
  size_t off = 0;
  auto carve = [&](size_t bytes) {
    void* p = ws + off;
    off = (off + bytes + 255) & ~(size_t)255;
    return p;
  };
  unsigned short* in_bf    = (unsigned short*)carve((size_t)M_TOT * DIN * 2);
  unsigned short* wkqv_bf  = (unsigned short*)carve((size_t)DIN * N_KQV * 2);
  unsigned short* wp_bf    = (unsigned short*)carve((size_t)DIN * N_P * 2);
  unsigned short* wproj_bf = (unsigned short*)carve((size_t)DIN * N_PROJ * 2);
  float*          kqv_f    = (float*)carve((size_t)M_TOT * N_KQV * 4);
  float*          p_f      = (float*)carve((size_t)M_TOT * N_P * 4);
  unsigned short* attn_bf  = (unsigned short*)carve((size_t)M_TOT * DIN * 2);
  const size_t CB = (size_t)NCHUNK * BH_TOT;         // chunk*bh slots
  float* A_Dg  = (float*)carve(CB * E_DIM * 4);
  float* A_Xs  = (float*)carve(CB * E_DIM * 4);
  float* A_Xk  = (float*)carve(CB * R_DIM * E_DIM * 4);
  float* A_Db  = (float*)carve(CB * 256 * 4);
  float* A_Xv  = (float*)carve(CB * 256 * 4);
  float* Cin_k = (float*)carve(CB * R_DIM * E_DIM * 4);
  float* Cin_s = (float*)carve(CB * E_DIM * 4);
  float* Cin_v = (float*)carve(CB * 256 * 4);

  // d_out carve-up (return-order concat, all fp32)
  float* out      = (float*)d_out;
  float* out_attn = out;                              // T*B*DIN
  float* out_fk   = out_attn + (size_t)M_TOT * DIN;   // B*R*H*E = 65536
  float* out_fv   = out_fk + (size_t)B_DIM * R_DIM * H_DIM * E_DIM;   // 16384
  float* out_fs   = out_fv + (size_t)B_DIM * R_DIM * H_DIM * HD_DIM;  // 16384
  float* out_tick = out_fs + (size_t)B_DIM * H_DIM * E_DIM;           // 8

  // 1) fp32 -> bf16 operand conversion
  cvt_f32_to_bf16<<<4096, 256, 0, stream>>>(inputs, in_bf, M_TOT * DIN);
  cvt_f32_to_bf16<<<4096, 256, 0, stream>>>(Wkqv, wkqv_bf, DIN * N_KQV);
  cvt_f32_to_bf16<<<256, 256, 0, stream>>>(Wp, wp_bf, DIN * N_P);
  cvt_f32_to_bf16<<<1024, 256, 0, stream>>>(Wproj, wproj_bf, DIN * N_PROJ);

  // 2) kqv = inputs @ W_kqv + b_kqv   (tiled WMMA, async-LDS-staged B)
  gemm_bf16_wmma_tiled<<<dim3(N_KQV / 64, M_TOT / 128), 256, 0, stream>>>(
      in_bf, wkqv_bf, bkqv, kqv_f, M_TOT, N_KQV, DIN);

  // 3) p = inputs @ W_p + b_p  (N=96 -> simple 16-wide WMMA path)
  gemm_bf16_wmma_small<<<dim3(N_P / 16, M_TOT / 128), 256, 0, stream>>>(
      in_bf, wp_bf, bp, p_f, M_TOT, N_P, DIN);

  // 4) chunked parallel scan + attention core
  scan_phaseA<<<dim3(NCHUNK, BH_TOT), 256, 0, stream>>>(
      kqv_f, p_f, term, tick, A_Dg, A_Xs, A_Xk, A_Db, A_Xv);
  scan_phaseB<<<BH_TOT, 256, 0, stream>>>(
      A_Dg, A_Xs, A_Xk, A_Db, A_Xv, tkprev, tvprev, sprev, tick,
      Cin_k, Cin_s, Cin_v, out_fk, out_fv, out_fs, out_tick);
  scan_phaseC<<<dim3(NCHUNK, BH_TOT), 256, 0, stream>>>(
      kqv_f, p_f, term, tick, Cin_k, Cin_s, Cin_v, attn_bf);

  // 5) attn_out = attn_pre @ W_proj + b_proj   (tiled WMMA)
  gemm_bf16_wmma_tiled<<<dim3(N_PROJ / 64, M_TOT / 128), 256, 0, stream>>>(
      attn_bf, wproj_bf, bproj, out_attn, M_TOT, N_PROJ, DIN);
}
